// GNNEstimator_9122510537263
// MI455X (gfx1250) — matrix operand order repaired and checked by reading, not compiled, not run on hardware
//
#include <hip/hip_runtime.h>
#include <hip/hip_bf16.h>
#include <math.h>

// ---------------- problem constants ----------------
#define NN 20000
#define EE 60000
#define HH 4
#define CC 128
#define DD 512
#define BB 4
#define SS 12
#define EP (EE + NN)     // edges + self loops = 80000
#define BN (BB * NN)     // 80000
#define BE (BB * EE)     // 240000

typedef __attribute__((ext_vector_type(16))) _Float16 v16h;
typedef __attribute__((ext_vector_type(8)))  float    v8f;

// ================= CDNA5 async copy (memory -> LDS, ASYNCcnt tracked) =================
__device__ __forceinline__ void async_b128(void* lds_ptr, const void* g_ptr) {
  unsigned int l = (unsigned int)(uintptr_t)lds_ptr;   // LDS aperture: addr[31:0] is the LDS offset
  asm volatile("global_load_async_to_lds_b128 %0, %1, off"
               :: "v"(l), "v"(g_ptr) : "memory");
}
__device__ __forceinline__ void wait_async0() {
  asm volatile("s_wait_asynccnt 0x0" ::: "memory");
}

// ================= WMMA fragment loaders (layouts per CDNA5 ISA) =================

// A-matrix 16x32 f16 fragment from a half tile (LDS or global), row-major, leading dim lda.
// lanes 0-15: row M=lane, K = {0..7, 16..23}; lanes 16-31: row M=lane-16, K = {8..15, 24..31}
__device__ __forceinline__ v16h load_a_frag_h(const _Float16* __restrict__ A, int lda,
                                              int k_base, int lane) {
  int m = lane & 15;
  int koff = (lane & 16) ? 8 : 0;
  const _Float16* r = A + m * lda + k_base + koff;
  v16h a;
#pragma unroll
  for (int i = 0; i < 8; ++i) a[i] = r[i];
#pragma unroll
  for (int i = 0; i < 8; ++i) a[8 + i] = r[16 + i];
  return a;
}

// B-matrix 32x16 f16 fragment from pre-transposed weights WT[Nc][K] (half).
// lanes 0-15: col N=lane, K=k_base+{0..7, 16..23}; lanes 16-31: col N=lane-16, K=k_base+{8..15, 24..31}
__device__ __forceinline__ v16h load_b_frag(const _Float16* __restrict__ WT, int ldk,
                                            int n_base, int k_base, int lane) {
  int n = lane & 15;
  int koff = (lane & 16) ? 8 : 0;
  const _Float16* p = WT + (size_t)(n_base + n) * (size_t)ldk + k_base + koff;
  v16h b;
#pragma unroll
  for (int i = 0; i < 8; ++i) b[i] = p[i];
#pragma unroll
  for (int i = 0; i < 8; ++i) b[8 + i] = p[16 + i];
  return b;
}

__device__ __forceinline__ v8f wmma_f16(v16h a, v16h b, v8f c) {
  return __builtin_amdgcn_wmma_f32_16x16x32_f16(false, a, false, b, (short)0, c, false, false);
}

// ================= small helper kernels =================

__global__ void k_fill(float* __restrict__ p, float v, long n) {
  long i = (long)blockIdx.x * blockDim.x + threadIdx.x;
  long stride = (long)gridDim.x * blockDim.x;
  for (; i < n; i += stride) p[i] = v;
}

// WT[n][k] = (half) W[k][n]   (W row-major [K][Nc])
__global__ void k_transpose_h(const float* __restrict__ W, _Float16* __restrict__ WT,
                              int K, int Nc) {
  long idx = (long)blockIdx.x * blockDim.x + threadIdx.x;
  long total = (long)K * Nc;
  if (idx >= total) return;
  int n = (int)(idx / K);
  int k = (int)(idx % K);
  WT[idx] = (_Float16)W[(size_t)k * Nc + n];
}

// x16[bn][c] = (f16)(nan0(vm)*W_in[0][c] + nan0(pb)*W_in[1][c] + nan0(qb)*W_in[2][c] + b_in[c])
__global__ void k_embed(const float* __restrict__ vm, const float* __restrict__ pb,
                        const float* __restrict__ qb, const float* __restrict__ W_in,
                        const float* __restrict__ b_in, _Float16* __restrict__ x16) {
  long gid = (long)blockIdx.x * blockDim.x + threadIdx.x;
  if (gid >= (long)BN * CC) return;
  int c = (int)(gid & (CC - 1));
  long bn = gid >> 7;
  int b = (int)(bn / NN);
  int n = (int)(bn % NN);
  long base = ((long)b * SS + (SS - 1)) * NN + n;
  float v0 = vm[base]; v0 = (v0 == v0) ? v0 : 0.f;
  float v1 = pb[base]; v1 = (v1 == v1) ? v1 : 0.f;
  float v2 = qb[base]; v2 = (v2 == v2) ? v2 : 0.f;
  x16[gid] = (_Float16)(v0 * W_in[c] + v1 * W_in[CC + c] + v2 * W_in[2 * CC + c] + b_in[c]);
}

// ======= LDS-staged WMMA GEMM: Y_f32[M,Nc] = act(A_f16[M,K] @ WT_f16[Nc,K] + bias) =======
// Block = MT m-tiles (16 rows each, contiguous in memory -> async-copied to LDS once),
// 8 waves split MT*ntiles output tiles.
template <int MT, bool RELU>
__global__ __launch_bounds__(256) void k_gemm(const _Float16* __restrict__ A,
                                              const _Float16* __restrict__ WT,
                                              const float* __restrict__ bias,
                                              float* __restrict__ Y, int K, int Nc) {
  __shared__ _Float16 sA[MT * 16 * 512];
  int tid = threadIdx.x, lane = tid & 31, wid = tid >> 5;
  long rowBase = (long)blockIdx.x * (MT * 16);

  // async stage: MT*16 consecutive rows = one contiguous block of MT*16*K halves
  const _Float16* src = A + rowBase * K;
  int chunks = MT * 16 * K / 8;          // 16-byte chunks
  for (int i = tid; i < chunks; i += 256)
    async_b128(&sA[(size_t)i * 8], src + (size_t)i * 8);
  wait_async0();
  __syncthreads();

  int ntiles = Nc >> 4;
  int nk = K >> 5;
  for (int w = wid; w < MT * ntiles; w += 8) {       // wave-uniform
    int s = w / ntiles, nt = w % ntiles;
    const _Float16* At = sA + s * 16 * K;
    v8f acc = {0.f, 0.f, 0.f, 0.f, 0.f, 0.f, 0.f, 0.f};
    for (int kt = 0; kt < nk; ++kt) {
      if (kt + 1 < nk)
        __builtin_prefetch(WT + (size_t)(nt * 16 + (lane & 15)) * K + (kt + 1) * 32, 0, 0);
      v16h a = load_a_frag_h(At, K, kt * 32, lane);
      v16h b = load_b_frag(WT, K, nt * 16, kt * 32, lane);
      acc = wmma_f16(a, b, acc);
    }
    int col = nt * 16 + (lane & 15);
    float bv = bias ? bias[col] : 0.f;
    long row0 = rowBase + s * 16 + ((lane & 16) ? 8 : 0);
#pragma unroll
    for (int r = 0; r < 8; ++r) {
      float v = acc[r] + bv;
      if (RELU) v = v > 0.f ? v : 0.f;
      Y[(size_t)(row0 + r) * Nc + col] = v;
    }
  }
}

// ================= GAT attention path =================

__global__ void k_attn_coef(const float* __restrict__ xh, const float* __restrict__ a_s,
                            const float* __restrict__ a_d, float* __restrict__ es,
                            float* __restrict__ ed) {
  long gid = (long)blockIdx.x * blockDim.x + threadIdx.x;
  if (gid >= (long)BN * HH) return;
  int h = (int)(gid & 3);
  long bn = gid >> 2;
  const float* row = xh + bn * DD + h * CC;
  const float* as = a_s + h * CC;
  const float* ad = a_d + h * CC;
  float s = 0.f, d = 0.f;
#pragma unroll 4
  for (int c = 0; c < CC; ++c) { float v = row[c]; s += v * as[c]; d += v * ad[c]; }
  es[gid] = s;
  ed[gid] = d;
}

__device__ __forceinline__ void atomicMaxF(float* a, float v) {
  if (v >= 0.f) atomicMax((int*)a, __float_as_int(v));
  else          atomicMin((unsigned int*)a, __float_as_uint(v));
}

__global__ void k_edge_logits(const float* __restrict__ es, const float* __restrict__ ed,
                              const int* __restrict__ ei0, const int* __restrict__ ei1,
                              float* __restrict__ ebuf, float* __restrict__ m) {
  long gid = (long)blockIdx.x * blockDim.x + threadIdx.x;
  if (gid >= (long)BB * EP * HH) return;
  int h = (int)(gid & 3);
  long t = gid >> 2;
  int e = (int)(t % EP);
  int b = (int)(t / EP);
  int src = (e < EE) ? ei0[e] : (e - EE);
  int dst = (e < EE) ? ei1[e] : (e - EE);
  float el = es[((long)b * NN + src) * HH + h] + ed[((long)b * NN + dst) * HH + h];
  el = (el > 0.f) ? el : 0.2f * el;
  ebuf[gid] = el;
  atomicMaxF(&m[((long)b * NN + dst) * HH + h], el);
}

__global__ void k_edge_exp(float* __restrict__ ebuf, const float* __restrict__ m,
                           const int* __restrict__ ei1, float* __restrict__ den) {
  long gid = (long)blockIdx.x * blockDim.x + threadIdx.x;
  if (gid >= (long)BB * EP * HH) return;
  int h = (int)(gid & 3);
  long t = gid >> 2;
  int e = (int)(t % EP);
  int b = (int)(t / EP);
  int dst = (e < EE) ? ei1[e] : (e - EE);
  float ex = expf(ebuf[gid] - m[((long)b * NN + dst) * HH + h]);
  ebuf[gid] = ex;
  atomicAdd(&den[((long)b * NN + dst) * HH + h], ex);
}

__global__ void k_aggregate(const float* __restrict__ ebuf, const float* __restrict__ den,
                            const int* __restrict__ ei0, const int* __restrict__ ei1,
                            const float* __restrict__ xh, float* __restrict__ aggr) {
  long bid = blockIdx.x;
  int e = (int)(bid % EP);
  int b = (int)(bid / EP);
  int src = (e < EE) ? ei0[e] : (e - EE);
  int dst = (e < EE) ? ei1[e] : (e - EE);
  int c = threadIdx.x;
  long srow = ((long)b * NN + src) * DD;
  long drow = ((long)b * NN + dst) * DD;
#pragma unroll
  for (int h = 0; h < HH; ++h) {
    float ex = ebuf[bid * HH + h];
    float dn = den[((long)b * NN + dst) * HH + h];
    float alpha = ex / (dn + 1e-16f);
    atomicAdd(&aggr[drow + h * CC + c], alpha * xh[srow + h * CC + c]);
  }
}

// h16 = (f16) relu(aggr + bias)   -- f16 because only GEMMs consume h
__global__ void k_bias_relu(const float* __restrict__ aggr, const float* __restrict__ bias,
                            _Float16* __restrict__ h16) {
  long gid = (long)blockIdx.x * blockDim.x + threadIdx.x;
  if (gid >= (long)BN * DD) return;
  float v = aggr[gid] + bias[(int)(gid & (DD - 1))];
  h16[gid] = (_Float16)(v > 0.f ? v : 0.f);
}

// ================= node head tail =================
__global__ void k_node_out(const float* __restrict__ s1, const float* __restrict__ sW2,
                           const float* __restrict__ sb2, float* __restrict__ out) {
  long bn = (long)blockIdx.x * blockDim.x + threadIdx.x;
  if (bn >= BN) return;
  const float* row = s1 + bn * 64;
  float a0 = sb2[0], a1 = sb2[1];
#pragma unroll 8
  for (int k = 0; k < 64; ++k) { float v = row[k]; a0 += v * sW2[k * 2]; a1 += v * sW2[k * 2 + 1]; }
  out[bn]      = 1.f / (1.f + expf(-a0)) * 0.3f + 0.85f;
  out[BN + bn] = tanhf(a1) * 0.5f;
}

// ================= fused edge head =================
// Per 16-edge tile: async-gather h2[src]/h2[dst] f16 rows -> LDS; ef = Asrc@pWt + Adst@pWb + pb
// (WMMA, into LDS); t = relu(ef@hW1 + hb1) (WMMA, LDS); p = t@hW2 + hb2; sigmoid -> d_out.
__global__ __launch_bounds__(256) void k_edge_head(
    const _Float16* __restrict__ h2, const int* __restrict__ ei0, const int* __restrict__ ei1,
    const _Float16* __restrict__ pWtT, const _Float16* __restrict__ pWbT,
    const float* __restrict__ pb, const _Float16* __restrict__ hW1T,
    const float* __restrict__ hb1, const float* __restrict__ hW2,
    const float* __restrict__ hb2, float* __restrict__ out) {
  __shared__ _Float16 sAll[32 * DD];   // rows 0..15 = src, 16..31 = dst  (32 KB)
  __shared__ _Float16 sEF[16 * DD];    // ef tile (16 KB)
  __shared__ _Float16 sT[16 * 64];
  __shared__ int sIdx[32];

  int tid  = threadIdx.x;
  int lane = tid & 31;
  int wid  = tid >> 5;
  long eg0 = (long)blockIdx.x * 16;
  int b   = (int)(eg0 / EE);           // 16 | EE, so a tile never crosses batches
  int el0 = (int)(eg0 % EE);

  if (tid < 16)       sIdx[tid] = ei0[el0 + tid];
  else if (tid < 32)  sIdx[tid] = ei1[el0 + (tid - 16)];
  __syncthreads();

  // async-gather 32 f16 rows (1 KB each) into LDS
  long nodeBase = (long)b * NN;
  for (int i = tid; i < 32 * (DD / 8); i += 256) {   // 16-byte chunks
    int row = i / (DD / 8);
    int c8  = i % (DD / 8);
    const _Float16* g = h2 + (nodeBase + sIdx[row]) * DD + c8 * 8;
    async_b128(&sAll[row * DD + c8 * 8], g);
  }
  wait_async0();
  __syncthreads();
  const _Float16* sA  = sAll;
  const _Float16* sBt = sAll + 16 * DD;

  // ef = Asrc @ pW_top^T + Adst @ pW_bot^T + pb   (Nc=512 -> 32 n-tiles over 8 waves)
#pragma unroll
  for (int j = 0; j < 4; ++j) {
    int nt = wid * 4 + j;
    v8f acc = {0.f, 0.f, 0.f, 0.f, 0.f, 0.f, 0.f, 0.f};
    for (int kt = 0; kt < DD / 32; ++kt) {
      v16h aS = load_a_frag_h(sA,  DD, kt * 32, lane);
      v16h aD = load_a_frag_h(sBt, DD, kt * 32, lane);
      v16h bS = load_b_frag(pWtT, DD, nt * 16, kt * 32, lane);
      v16h bD = load_b_frag(pWbT, DD, nt * 16, kt * 32, lane);
      acc = wmma_f16(aS, bS, acc);
      acc = wmma_f16(aD, bD, acc);
    }
    int col = nt * 16 + (lane & 15);
    float bv = pb[col];
    int row0 = (lane & 16) ? 8 : 0;
#pragma unroll
    for (int r = 0; r < 8; ++r) sEF[(row0 + r) * DD + col] = (_Float16)(acc[r] + bv);
  }
  __syncthreads();

  // t = relu(ef @ hW1 + hb1)   (Nc=64 -> 4 n-tiles, waves 0..3)
  if (wid < 4) {
    int nt = wid;
    v8f acc = {0.f, 0.f, 0.f, 0.f, 0.f, 0.f, 0.f, 0.f};
    for (int kt = 0; kt < DD / 32; ++kt) {
      v16h a  = load_a_frag_h(sEF, DD, kt * 32, lane);
      v16h bb = load_b_frag(hW1T, DD, nt * 16, kt * 32, lane);
      acc = wmma_f16(a, bb, acc);
    }
    int col = nt * 16 + (lane & 15);
    float bv = hb1[col];
    int row0 = (lane & 16) ? 8 : 0;
#pragma unroll
    for (int r = 0; r < 8; ++r) {
      float v = acc[r] + bv;
      sT[(row0 + r) * 64 + col] = (_Float16)(v > 0.f ? v : 0.f);
    }
  }
  __syncthreads();

  // p = t @ hW2 + hb2 ; sigmoid*2.99+0.01 -> r_line / x_line
  if (tid < 32) {
    int er = tid >> 1, j = tid & 1;
    float s = hb2[j];
#pragma unroll 8
    for (int k = 0; k < 64; ++k) s += (float)sT[er * 64 + k] * hW2[k * 2 + j];
    float val = 1.f / (1.f + expf(-s)) * 2.99f + 0.01f;
    out[2L * BN + (long)j * BE + eg0 + er] = val;
  }
}

// ================= host-side orchestration =================

static inline unsigned nblk(long n, int t) { return (unsigned)((n + t - 1) / t); }

extern "C" void kernel_launch(void* const* d_in, const int* in_sizes, int n_in,
                              void* d_out, int out_size, void* d_ws, size_t ws_size,
                              hipStream_t stream) {
  const float* v_mag = (const float*)d_in[0];
  const float* p_bus = (const float*)d_in[1];
  const float* q_bus = (const float*)d_in[2];
  const int*   ei    = (const int*)d_in[3];
  const int*   ei0   = ei;
  const int*   ei1   = ei + EE;
  const float* W_in  = (const float*)d_in[4];
  const float* b_in  = (const float*)d_in[5];
  const float* g1_W  = (const float*)d_in[6];
  const float* g1_as = (const float*)d_in[7];
  const float* g1_ad = (const float*)d_in[8];
  const float* g1_b  = (const float*)d_in[9];
  const float* g2_W  = (const float*)d_in[10];
  const float* g2_as = (const float*)d_in[11];
  const float* g2_ad = (const float*)d_in[12];
  const float* g2_b  = (const float*)d_in[13];
  const float* sW1   = (const float*)d_in[14];
  const float* sb1   = (const float*)d_in[15];
  const float* sW2   = (const float*)d_in[16];
  const float* sb2   = (const float*)d_in[17];
  const float* pW    = (const float*)d_in[18];
  const float* pb    = (const float*)d_in[19];
  const float* hW1   = (const float*)d_in[20];
  const float* hb1   = (const float*)d_in[21];
  const float* hW2   = (const float*)d_in[22];
  const float* hb2   = (const float*)d_in[23];
  float* out = (float*)d_out;

  // ---- carve workspace ----
  char* base = (char*)d_ws;
  size_t off = 0;
  auto carve = [&](size_t bytes) -> void* {
    void* p = base + off;
    off += (bytes + 255) & ~(size_t)255;
    return p;
  };
  _Float16* g1WT = (_Float16*)carve((size_t)DD * CC * 2);
  _Float16* g2WT = (_Float16*)carve((size_t)DD * DD * 2);
  _Float16* sW1T = (_Float16*)carve((size_t)64 * DD * 2);
  _Float16* pWtT = (_Float16*)carve((size_t)DD * DD * 2);
  _Float16* pWbT = (_Float16*)carve((size_t)DD * DD * 2);
  _Float16* hW1T = (_Float16*)carve((size_t)64 * DD * 2);
  _Float16* x16  = (_Float16*)carve((size_t)BN * CC * 2);  // f16 embeddings
  _Float16* h16  = (_Float16*)carve((size_t)BN * DD * 2);  // f16 h1 / h2 (GEMM inputs)
  float* bufA = (float*)carve((size_t)BN * DD * 4);        // xh (f32, attention/aggregation)
  float* bufB = (float*)carve((size_t)BN * DD * 4);        // aggregation accumulator (f32)
  float* s1   = (float*)carve((size_t)BN * 64 * 4);        // node-head hidden
  float* es   = (float*)carve((size_t)BN * HH * 4);
  float* ed   = (float*)carve((size_t)BN * HH * 4);
  float* mseg = (float*)carve((size_t)BN * HH * 4);
  float* den  = (float*)carve((size_t)BN * HH * 4);
  float* ebuf = (float*)carve((size_t)BB * EP * HH * 4);
  (void)ws_size; (void)in_sizes; (void)n_in; (void)out_size;

  // ---- weight prep: f32 -> f16, transposed to [Nc][K] ----
  k_transpose_h<<<nblk((long)CC * DD, 256), 256, 0, stream>>>(g1_W, g1WT, CC, DD);
  k_transpose_h<<<nblk((long)DD * DD, 256), 256, 0, stream>>>(g2_W, g2WT, DD, DD);
  k_transpose_h<<<nblk((long)DD * 64, 256), 256, 0, stream>>>(sW1, sW1T, DD, 64);
  k_transpose_h<<<nblk((long)DD * DD, 256), 256, 0, stream>>>(pW, pWtT, DD, DD);
  k_transpose_h<<<nblk((long)DD * DD, 256), 256, 0, stream>>>(pW + (size_t)DD * DD, pWbT, DD, DD);
  k_transpose_h<<<nblk((long)DD * 64, 256), 256, 0, stream>>>(hW1, hW1T, DD, 64);

  // ---- input embedding (f16) ----
  k_embed<<<nblk((long)BN * CC, 256), 256, 0, stream>>>(v_mag, p_bus, q_bus, W_in, b_in, x16);

  // ---- GAT layer 1: xh1 = x @ g1_W ----
  k_gemm<1, false><<<BN / 16, 256, 0, stream>>>(x16, g1WT, nullptr, bufA, CC, DD);
  k_attn_coef<<<nblk((long)BN * HH, 256), 256, 0, stream>>>(bufA, g1_as, g1_ad, es, ed);
  k_fill<<<1024, 256, 0, stream>>>(mseg, -1e30f, (long)BN * HH);
  k_fill<<<1024, 256, 0, stream>>>(den, 0.f, (long)BN * HH);
  k_fill<<<4096, 256, 0, stream>>>(bufB, 0.f, (long)BN * DD);
  k_edge_logits<<<nblk((long)BB * EP * HH, 256), 256, 0, stream>>>(es, ed, ei0, ei1, ebuf, mseg);
  k_edge_exp<<<nblk((long)BB * EP * HH, 256), 256, 0, stream>>>(ebuf, mseg, ei1, den);
  k_aggregate<<<(unsigned)((long)BB * EP), CC, 0, stream>>>(ebuf, den, ei0, ei1, bufA, bufB);
  k_bias_relu<<<nblk((long)BN * DD, 256), 256, 0, stream>>>(bufB, g1_b, h16);   // h1 (f16)

  // ---- GAT layer 2: xh2 = h1 @ g2_W ----
  k_gemm<1, false><<<BN / 16, 256, 0, stream>>>(h16, g2WT, nullptr, bufA, DD, DD);
  k_attn_coef<<<nblk((long)BN * HH, 256), 256, 0, stream>>>(bufA, g2_as, g2_ad, es, ed);
  k_fill<<<1024, 256, 0, stream>>>(mseg, -1e30f, (long)BN * HH);
  k_fill<<<1024, 256, 0, stream>>>(den, 0.f, (long)BN * HH);
  k_fill<<<4096, 256, 0, stream>>>(bufB, 0.f, (long)BN * DD);
  k_edge_logits<<<nblk((long)BB * EP * HH, 256), 256, 0, stream>>>(es, ed, ei0, ei1, ebuf, mseg);
  k_edge_exp<<<nblk((long)BB * EP * HH, 256), 256, 0, stream>>>(ebuf, mseg, ei1, den);
  k_aggregate<<<(unsigned)((long)BB * EP), CC, 0, stream>>>(ebuf, den, ei0, ei1, bufA, bufB);
  k_bias_relu<<<nblk((long)BN * DD, 256), 256, 0, stream>>>(bufB, g2_b, h16);   // h2 (f16)

  // ---- node head: s1 = relu(h2@sW1+sb1) [WMMA, 2 m-tiles/block], then 64->2 tail ----
  k_gemm<2, true><<<BN / 32, 256, 0, stream>>>(h16, sW1T, sb1, s1, DD, 64);
  k_node_out<<<nblk(BN, 256), 256, 0, stream>>>(s1, sW2, sb2, out);

  // ---- fused edge head (dominant GEMM chain, all WMMA, async-gathered LDS tiles) ----
  k_edge_head<<<(unsigned)(BE / 16), 256, 0, stream>>>(h16, ei0, ei1, pWtT, pWbT, pb,
                                                       hW1T, hb1, hW2, hb2, out);
}
